// CalculateFlow_45930380264076
// MI455X (gfx1250) — matrix operand order repaired and checked by reading, not compiled, not run on hardware
//
#include <hip/hip_runtime.h>

// ---------------------------------------------------------------------------
// Block-matching optical flow for MI455X (gfx1250, wave32, WMMA).
//
// Cost volume = 5x5 box filter of per-displacement |F-G| planes.
// Horizontal 5-tap SAD row sums via v_sad_u8 on DWORD-packed shifted windows
// (conflict-free LDS layout, stride 27 DWORDs); vertical 5-sum is ONE
// v_wmma_f32_16x16x32_f16 per displacement:
//   D = Sv(16x20 const band, A operand) x Hrow(20x16, B operand)
// Row sums <= 1275 are exact in f16; f32 accumulation -> bit-exact integer
// SAD costs, so spiral-tie-break argmin (cost*64 + rank <= 408048 < 2^24) is
// exact in f32. One wave per 16x16 tile, no barriers (per-wave LDS tiles,
// same-wave LDS ops are in-order on CDNA5).
// ---------------------------------------------------------------------------

#define HH 544
#define WW 960
#define HWSZ (HH * WW)
#define TILEX (WW / 16)              // 60
#define NTILES ((HH / 16) * TILEX)   // 34*60 = 2040, divisible by 8

#define FSTR 32                      // F byte tile row stride (26 used, zero pad)
#define WSTR 27                      // WF window tile row stride in DWORDs:
                                     // 16*27 mod 64 = 48 -> both lane halves
                                     // hit disjoint bank ranges (conflict-free)

typedef __attribute__((ext_vector_type(16))) _Float16 v16h;
typedef __attribute__((ext_vector_type(8)))  float    v8f;

// spiral argmin tie-break ranks, indexed d = jj*7 + ii (dy = jj-3, dx = ii-3)
__device__ __constant__ float SPIRAL_D[49] = {
  42,43,44,45,46,47,48,
  41,20,21,22,23,24,25,
  40,19, 6, 7, 8, 9,26,
  39,18, 5, 0, 1,10,27,
  38,17, 4, 3, 2,11,28,
  37,16,15,14,13,12,29,
  36,35,34,33,32,31,30};

__device__ __forceinline__ float ldc(const float* p, int y, int x) {
  y = min(max(y, 0), HH - 1);
  x = min(max(x, 0), WW - 1);
  return p[y * WW + x];
}

// ---- kernel 1: separable [1 2 1]/4 blur (replicate pad) + u8 quantize ------
__global__ __launch_bounds__(256)
void k_blur_quant(const float* __restrict__ f, const float* __restrict__ g,
                  float* __restrict__ fblur,
                  unsigned char* __restrict__ fq, unsigned char* __restrict__ gq) {
  int idx = blockIdx.x * 256 + threadIdx.x;
  if (idx >= HWSZ) return;
  int y = idx / WW, x = idx % WW;
  float vf[3], vg[3];
#pragma unroll
  for (int j = 0; j < 3; ++j) {            // vertical pass at cols x-1..x+1
    int xx = x + j - 1;                    // clamp inside ldc == replicate pad
    vf[j] = (ldc(f, y - 1, xx) + 2.0f * ldc(f, y, xx) + ldc(f, y + 1, xx)) / 4.0f;
    vg[j] = (ldc(g, y - 1, xx) + 2.0f * ldc(g, y, xx) + ldc(g, y + 1, xx)) / 4.0f;
  }
  float fb = (vf[0] + 2.0f * vf[1] + vf[2]) / 4.0f;
  float gb = (vg[0] + 2.0f * vg[1] + vg[2]) / 4.0f;
  fblur[idx] = fb;
  float fr = fminf(fmaxf(rintf(fb * 255.0f), 0.0f), 255.0f);  // rintf = RNE (jnp.round)
  float gr = fminf(fmaxf(rintf(gb * 255.0f), 0.0f), 255.0f);
  fq[idx] = (unsigned char)(int)fr;
  gq[idx] = (unsigned char)(int)gr;
}

// ---- kernel 2: central-difference gradient of f_blur (replicate pad) -------
__global__ __launch_bounds__(256)
void k_grad(const float* __restrict__ fb,
            float* __restrict__ dfx, float* __restrict__ dfy) {
  int idx = blockIdx.x * 256 + threadIdx.x;
  if (idx >= HWSZ) return;
  int y = idx / WW, x = idx % WW;
  dfy[idx] = (ldc(fb, y + 1, x) - ldc(fb, y - 1, x)) * 0.5f;
  dfx[idx] = (ldc(fb, y, x + 1) - ldc(fb, y, x - 1)) * 0.5f;
}

// ---- kernel 3: WMMA SAD cost volume + spiral argmin + LK subpixel ----------
__global__ __launch_bounds__(256)
void k_flow(const unsigned char* __restrict__ fq, const unsigned char* __restrict__ gq,
            const float* __restrict__ dfx_g, const float* __restrict__ dfy_g,
            float* __restrict__ flow) {
  // per-wave LDS (8 waves/block, ~7.2 KB each, ~58 KB total)
  __shared__ unsigned char sF[8][26 * FSTR];     // f_q bytes, halo 5, zero-fill
  __shared__ unsigned char sG[8][20 * 20];       // g_q bytes, halo 2, zero-fill
  __shared__ unsigned int  sWF[8][26 * WSTR];    // packed F windows: WF[r][c] = F[r][c..c+3]
  __shared__ float sDx[8][20 * 20];              // dfx, halo 2, zero-fill
  __shared__ float sDy[8][20 * 20];              // dfy, halo 2, zero-fill

  const int wid  = threadIdx.x >> 5;
  const int lane = threadIdx.x & 31;
  const int tile = blockIdx.x * 8 + wid;         // 2040 tiles exactly
  const int ty0  = (tile / TILEX) * 16;
  const int tx0  = (tile % TILEX) * 16;

  unsigned char* F  = sF[wid];
  unsigned char* G  = sG[wid];
  unsigned int*  WF = sWF[wid];
  float* DX = sDx[wid];
  float* DY = sDy[wid];

  for (int i = lane; i < 26 * FSTR; i += 32) {
    int r = i >> 5, c = i & (FSTR - 1);
    int y = ty0 - 5 + r, x = tx0 - 5 + c;
    F[i] = (c < 26 && y >= 0 && y < HH && x >= 0 && x < WW) ? fq[y * WW + x]
                                                            : (unsigned char)0;
  }
  for (int i = lane; i < 20 * 20; i += 32) {
    int r = i / 20, c = i % 20;
    int y = ty0 - 2 + r, x = tx0 - 2 + c;
    bool in = (y >= 0 && y < HH && x >= 0 && x < WW);
    int gi = in ? (y * WW + x) : 0;
    G[i]  = in ? gq[gi]    : (unsigned char)0;
    DX[i] = in ? dfx_g[gi] : 0.0f;
    DY[i] = in ? dfy_g[gi] : 0.0f;
  }
  // pack shifted 4-byte windows (one-time; same-wave LDS ops are in-order)
  for (int i = lane; i < 26 * 26; i += 32) {
    int r = i / 26, c = i % 26;
    const unsigned char* p = F + r * FSTR + c;   // c+3 <= 28 < FSTR (zero pad)
    WF[r * WSTR + c] = (unsigned)p[0] | ((unsigned)p[1] << 8) |
                       ((unsigned)p[2] << 16) | ((unsigned)p[3] << 24);
  }

  const bool lo = lane < 16;
  const int  N  = lane & 15;   // output column within tile; also A-matrix row M

  // A operand: constant band matrix Sv[i,k] = (i<=k<=i+4), K padded 20->32.
  // A k order used here (matches B): lane L<16 row M=L holds K=0..15;
  // lane L>=16 row M=L-16 holds K=16..31.
  v16h amat;
#pragma unroll
  for (int s = 0; s < 16; ++s) {
    int K = (lo ? 0 : 16) + s;
    amat[s] = (_Float16)((K >= N && K <= N + 4) ? 1.0f : 0.0f);
  }

  // G-side taps are displacement-invariant: hoist each lane's 16 packed
  // windows (4 bytes + 5th byte) into registers once.
  unsigned int Gw[16], G5[16];
#pragma unroll
  for (int s = 0; s < 16; ++s) {
    int K = lo ? s : 16 + s;
    unsigned int w = 0u, b5 = 0u;
    if (K < 20) {
      const unsigned char* gr = G + K * 20 + N;
      w  = (unsigned)gr[0] | ((unsigned)gr[1] << 8) |
           ((unsigned)gr[2] << 16) | ((unsigned)gr[3] << 24);
      b5 = (unsigned)gr[4];
    }
    Gw[s] = w;
    G5[s] = b5;
  }

  float bestC[8];
  int   bestD[8];
#pragma unroll
  for (int r = 0; r < 8; ++r) { bestC[r] = 3.0e38f; bestD[r] = 0; }

#pragma unroll 1
  for (int d = 0; d < 49; ++d) {
    const int jj = d / 7, ii = d % 7;     // dy = jj-3, dx = ii-3
    // B operand: Hrow[k][n] = 5-tap horizontal SAD, rows 20..31 zero.
    // k order used here (same as A): lanes 0-15 K=0..15, lanes 16-31 K=16..31,
    // VGPR v packs K=2v (lo16), K=2v+1 (hi16).
    v16h bmat;
#pragma unroll
    for (int s = 0; s < 16; ++s) {
      int K = lo ? s : 16 + s;
      unsigned int hs = 0u;
      if (K < 20) {
        int rb = (K + jj) * WSTR + (N + ii);
        unsigned int fw0 = WF[rb];                 // conflict-free ds_load_b32
        unsigned int fw4 = WF[rb + 4] & 0xFFu;     // 5th byte only
        hs = __builtin_amdgcn_sad_u8(fw0, Gw[s], 0u);      // v_sad_u8: 4 taps
        hs = __builtin_amdgcn_sad_u8(fw4, G5[s], hs);      // + 5th tap
      }
      bmat[s] = (_Float16)(unsigned short)hs;      // <= 1275, exact in f16
    }
    v8f cacc = {};
    // D[i][n] = sum_k Sv[i][k] * Hrow[k][n] == 5x5 SAD cost, exact int in f32
    cacc = __builtin_amdgcn_wmma_f32_16x16x32_f16(
        false, amat, false, bmat, (short)0, cacc, false, false);
    const float rk = SPIRAL_D[d];
#pragma unroll
    for (int r = 0; r < 8; ++r) {         // D layout: VGPR r -> M = 8*(lane/16)+r
      float comb = cacc[r] * 64.0f + rk;  // exact: <= 408048 < 2^24
      if (comb < bestC[r]) { bestC[r] = comb; bestD[r] = d; }
    }
  }

  // LK subpixel on the 16-tap template ring + flow write, per-lane pixels
  constexpr int RTY[16] = {-2,-2,-2,-2,-2,-1,-1, 0, 0, 1, 1, 2, 2, 2, 2, 2};
  constexpr int RTX[16] = {-2,-1, 0, 1, 2,-2, 2,-2, 2,-2, 2,-2,-1, 0, 1, 2};
#pragma unroll 1
  for (int r = 0; r < 8; ++r) {
    const int M  = (lo ? 0 : 8) + r;
    const int y  = ty0 + M, x = tx0 + N;
    const int d  = bestD[r];
    const int jj = d / 7, ii = d % 7;
    float aa = 0.0f, bb = 0.0f, dd = 0.0f, pp = 0.0f, qq = 0.0f;
#pragma unroll
    for (int t = 0; t < 16; ++t) {
      int gi = (M + 2 + RTY[t]) * 20 + (N + 2 + RTX[t]);
      float gx = DX[gi], gy = DY[gi];
      float gv = (float)G[gi] / 255.0f;
      float fv = (float)F[(M + jj + RTY[t] + 2) * FSTR + (N + ii + RTX[t] + 2)] / 255.0f;
      float z  = gv - fv;
      aa += gx * gx; bb += gx * gy; dd += gy * gy;
      pp += z * gx;  qq += z * gy;
    }
    float det = aa * dd - bb * bb;
    float su = 0.0f, sv = 0.0f;
    if (det > 1e-7f) {
      su = (dd * pp - bb * qq) / det;
      sv = (aa * qq - bb * pp) / det;
      if (fabsf(su) >= 1.0f) su = 0.0f;
      if (fabsf(sv) >= 1.0f) sv = 0.0f;
    }
    flow[0 * HWSZ + y * WW + x] = (float)(3 - jj) + sv;   // v (dy) channel
    flow[1 * HWSZ + y * WW + x] = (float)(3 - ii) + su;   // u (dx) channel
  }
}

// ---- kernel 4: 3x3 median (replicate pad), 19-op min/max network -----------
__device__ __forceinline__ void mnmx(float& a, float& b) {
  float t = fminf(a, b);
  b = fmaxf(a, b);
  a = t;
}

__global__ __launch_bounds__(256)
void k_median(const float* __restrict__ flow, float* __restrict__ out) {
  int idx = blockIdx.x * 256 + threadIdx.x;
  if (idx >= 2 * HWSZ) return;
  int ch = idx / HWSZ;
  int p  = idx % HWSZ;
  int y = p / WW, x = p % WW;
  const float* src = flow + ch * HWSZ;
  float v[9];
#pragma unroll
  for (int j = 0; j < 3; ++j)
#pragma unroll
    for (int i = 0; i < 3; ++i) {
      int yy = min(max(y + j - 1, 0), HH - 1);
      int xx = min(max(x + i - 1, 0), WW - 1);
      v[j * 3 + i] = src[yy * WW + xx];
    }
  mnmx(v[1], v[2]); mnmx(v[4], v[5]); mnmx(v[7], v[8]);
  mnmx(v[0], v[1]); mnmx(v[3], v[4]); mnmx(v[6], v[7]);
  mnmx(v[1], v[2]); mnmx(v[4], v[5]); mnmx(v[7], v[8]);
  mnmx(v[0], v[3]); mnmx(v[5], v[8]); mnmx(v[4], v[7]);
  mnmx(v[3], v[6]); mnmx(v[1], v[4]); mnmx(v[2], v[5]);
  mnmx(v[4], v[7]); mnmx(v[4], v[2]); mnmx(v[6], v[4]);
  mnmx(v[4], v[2]);
  out[idx] = v[4];
}

// ---------------------------------------------------------------------------
extern "C" void kernel_launch(void* const* d_in, const int* in_sizes, int n_in,
                              void* d_out, int out_size, void* d_ws, size_t ws_size,
                              hipStream_t stream) {
  (void)in_sizes; (void)n_in; (void)out_size; (void)ws_size;
  const float* f = (const float*)d_in[0];
  const float* g = (const float*)d_in[1];
  char* ws = (char*)d_ws;
  // workspace layout (22*HWSZ bytes ~= 11.5 MB total)
  float*         fblur = (float*)(ws);
  float*         dfx   = (float*)(ws + 4ull  * HWSZ);
  float*         dfy   = (float*)(ws + 8ull  * HWSZ);
  unsigned char* fq    = (unsigned char*)(ws + 12ull * HWSZ);
  unsigned char* gq    = (unsigned char*)(ws + 13ull * HWSZ);
  float*         flow  = (float*)(ws + 14ull * HWSZ);
  float*         out   = (float*)d_out;

  const int nb = (HWSZ + 255) / 256;
  k_blur_quant<<<nb, 256, 0, stream>>>(f, g, fblur, fq, gq);
  k_grad<<<nb, 256, 0, stream>>>(fblur, dfx, dfy);
  k_flow<<<NTILES / 8, 256, 0, stream>>>(fq, gq, dfx, dfy, flow);
  k_median<<<(2 * HWSZ + 255) / 256, 256, 0, stream>>>(flow, out);
}
